// SoftmaxAttention_30502857736672
// MI455X (gfx1250) — matrix operand order repaired and checked by reading, not compiled, not run on hardware
//
#include <hip/hip_runtime.h>

// Problem constants (from reference): B=64, La=Lb=512, D=512, fp32.
#define B_    64
#define L_    512
#define D_    512
#define TQ    64              // Q rows per workgroup
#define PQ    528             // bf16 elements per row (Q/P and K-stage buffers); 528*2=1056=33*32 -> 32B-aligned rows
#define SST   516             // f32 elements per row, S buffer
#define VTS   80              // bf16 elements per row of transposed V tile (160B = 5*32B)
#define NEGV  (-10000.0f)

typedef __bf16 bf16_t;
typedef bf16_t v16bf __attribute__((ext_vector_type(16)));
typedef bf16_t v8bf  __attribute__((ext_vector_type(8)));
typedef float  v8f   __attribute__((ext_vector_type(8)));

#define QS_BYTES   (TQ * PQ * 2)        // 67584  : Q tile bf16, reused as P after softmax
#define S_BYTES    (TQ * SST * 4)       // 132096 : scores f32
#define KV_BYTES   (D_ * VTS * 2)       // 81920  : K stage (phase A) / transposed V tile (phase C)
#define BIAS_BYTES (L_ * 4)             // 2048
#define SMEM_BYTES (QS_BYTES + S_BYTES + KV_BYTES + BIAS_BYTES)  // 283648 < 320KB

// A-matrix (16x32 bf16) fragment, contiguous k order (B fragments use the same):
// lane l<16 (row l)    holds K = k0 + {0..15}
// lane l>=16 (row l-16) holds K = k0 + {16..31}
static __device__ inline v16bf load_a_frag(const bf16_t* rowbase, int k0, int lrow, int h2) {
  const bf16_t* p = rowbase + lrow * PQ + k0 + h2 * 16;
  v8bf c0 = *(const v8bf*)(p);
  v8bf c1 = *(const v8bf*)(p + 8);
  return __builtin_shufflevector(c0, c1, 0,1,2,3,4,5,6,7,8,9,10,11,12,13,14,15);
}

// Cooperative 64x512 f32 -> bf16 row-major tile stage (256 threads, float4 coalesced).
static __device__ inline void stage_tile_bf16(const float* __restrict__ src,
                                              bf16_t* __restrict__ dst, int tid) {
#pragma unroll
  for (int i = 0; i < 32; ++i) {
    int e4 = i * 256 + tid;                 // float4 index within 64x512
    int r  = e4 >> 7;                       // row 0..63
    int d0 = (e4 & 127) * 4;                // col
    float4 v = *(const float4*)(src + (size_t)r * D_ + d0);
    bf16_t* d = dst + r * PQ + d0;
    d[0] = (bf16_t)v.x; d[1] = (bf16_t)v.y;
    d[2] = (bf16_t)v.z; d[3] = (bf16_t)v.w;
  }
}

// One attention direction: out[b, q, :] = softmax(Q Kv^T + bias(maskK)) Kv
__global__ void __launch_bounds__(256, 1)
attn_fused_kernel(const float* __restrict__ Qg, const float* __restrict__ KVg,
                  const float* __restrict__ maskK, float* __restrict__ outg) {
  extern __shared__ char smem[];
  bf16_t* qs    = (bf16_t*)smem;                               // Q tile; becomes P
  float*  sS    = (float*)(smem + QS_BYTES);                   // scores
  bf16_t* kv    = (bf16_t*)(smem + QS_BYTES + S_BYTES);        // K stage / V^T stage
  float*  biasK = (float*)(smem + QS_BYTES + S_BYTES + KV_BYTES);

  const int tid  = threadIdx.x;
  const int lane = tid & 31;
  const int wave = tid >> 5;     // 0..7
  const int mt   = wave & 3;     // M-tile (16 rows) within the 64-row Q tile
  const int hh   = wave >> 2;    // half selector (32-wide N slice / 256-wide D half)
  const int lrow = lane & 15;
  const int h2   = lane >> 4;
  const int rowb = mt * 16 + h2 * 8;

  const int b  = blockIdx.x >> 3;
  const int q0 = (blockIdx.x & 7) * TQ;

  const float* kvbatch = KVg + (size_t)b * L_ * D_;

  // ---------------- Phase 0: stage Q tile f32->bf16, mask bias ----------------
  stage_tile_bf16(Qg + ((size_t)b * L_ + q0) * D_, qs, tid);
  for (int n = tid; n < L_; n += 256)
    biasK[n] = (1.0f - maskK[(size_t)b * L_ + n]) * NEGV;
  __syncthreads();

  // ---------------- Phase A: S = Q Kv^T + bias (K staged through LDS) ----------------
  {
    const v8f zero = {0.f,0.f,0.f,0.f,0.f,0.f,0.f,0.f};
    const bf16_t* arow = qs + mt * 16 * PQ;
    for (int n0 = 0; n0 < L_; n0 += 64) {
      __syncthreads();
      stage_tile_bf16(kvbatch + (size_t)n0 * D_, kv, tid);
      if (n0 + 64 < L_)  // global_prefetch_b8 for next K block
        __builtin_prefetch(kvbatch + (size_t)(n0 + 64 + (tid >> 2)) * D_ + (tid & 3) * 128, 0, 1);
      __syncthreads();
      v8f acc0 = zero, acc1 = zero;
      // B fragment: lane l -> K row (n-col) hh*32 + j*16 + lrow, contiguous d chunk
      const bf16_t* b0 = kv + (hh * 32 + lrow) * PQ + h2 * 16;
      const bf16_t* b1 = b0 + 16 * PQ;
#pragma unroll 4
      for (int kd = 0; kd < 16; ++kd) {            // D = 16 * 32
        v16bf a = load_a_frag(arow, kd * 32, lrow, h2);
        v16bf bb0 = *(const v16bf*)(b0 + kd * 32);
        v16bf bb1 = *(const v16bf*)(b1 + kd * 32);
        acc0 = __builtin_amdgcn_wmma_f32_16x16x32_bf16(false, a, false, bb0, (short)0, acc0, false, false);
        acc1 = __builtin_amdgcn_wmma_f32_16x16x32_bf16(false, a, false, bb1, (short)0, acc1, false, false);
      }
      const int n_0 = n0 + hh * 32 + lrow;
      const float bias0 = biasK[n_0];
      const float bias1 = biasK[n_0 + 16];
#pragma unroll
      for (int r = 0; r < 8; ++r) {
        sS[(rowb + r) * SST + n_0]      = acc0[r] + bias0;
        sS[(rowb + r) * SST + n_0 + 16] = acc1[r] + bias1;
      }
    }
  }
  __syncthreads();

  // ---------------- Phase B: exact row softmax, P stored bf16 over Q ----------------
  {
    const int row = tid >> 2;          // 4 lanes per row
    const int seg = tid & 3;
    const float* srow = sS + row * SST + seg * 128;
    float m = -3.4e38f;
#pragma unroll 8
    for (int i = 0; i < 128; ++i) m = fmaxf(m, srow[i]);
    m = fmaxf(m, __shfl_xor(m, 1, 32));
    m = fmaxf(m, __shfl_xor(m, 2, 32));
    float s = 0.0f;
#pragma unroll 8
    for (int i = 0; i < 128; ++i) s += __expf(srow[i] - m);
    s += __shfl_xor(s, 1, 32);
    s += __shfl_xor(s, 2, 32);
    float inv = 1.0f / s;
    bf16_t* prow = qs + row * PQ + seg * 128;
#pragma unroll 8
    for (int i = 0; i < 128; ++i) prow[i] = (bf16_t)(__expf(srow[i] - m) * inv);
  }

  // ---------------- Phase C: O = P V (V staged transposed bf16 in LDS) ----------------
  v8f accO[16];
  {
    const v8f zero = {0.f,0.f,0.f,0.f,0.f,0.f,0.f,0.f};
#pragma unroll
    for (int i = 0; i < 16; ++i) accO[i] = zero;
  }
  for (int kk0 = 0; kk0 < L_; kk0 += 64) {
    __syncthreads();
#pragma unroll
    for (int i = 0; i < 32; ++i) {
      int e4 = i * 256 + tid;
      int r  = e4 >> 7;                       // V row within tile
      int d0 = (e4 & 127) * 4;                // V column
      float4 v = *(const float4*)(kvbatch + (size_t)(kk0 + r) * D_ + d0);
      kv[(d0 + 0) * VTS + r] = (bf16_t)v.x;   // transpose: Vt[d][k]
      kv[(d0 + 1) * VTS + r] = (bf16_t)v.y;
      kv[(d0 + 2) * VTS + r] = (bf16_t)v.z;
      kv[(d0 + 3) * VTS + r] = (bf16_t)v.w;
    }
    if (kk0 + 64 < L_)  // global_prefetch_b8 for next V tile
      __builtin_prefetch(kvbatch + (size_t)(kk0 + 64 + (tid >> 2)) * D_ + (tid & 3) * 128, 0, 1);
    __syncthreads();
    const bf16_t* prow = qs + mt * 16 * PQ;
#pragma unroll
    for (int kl = 0; kl < 64; kl += 32) {
      v16bf a = load_a_frag(prow, kk0 + kl, lrow, h2);
#pragma unroll
      for (int dt = 0; dt < 16; ++dt) {
        const v16bf* bp = (const v16bf*)(kv + (hh * 256 + dt * 16 + lrow) * VTS
                                         + kl + h2 * 16);
        accO[dt] = __builtin_amdgcn_wmma_f32_16x16x32_bf16(
            false, a, false, *bp, (short)0, accO[dt], false, false);
      }
    }
  }

  // ---------------- Write output ----------------
  {
    float* op = outg + ((size_t)b * L_ + q0) * D_;
#pragma unroll
    for (int dt = 0; dt < 16; ++dt) {
      int col = hh * 256 + dt * 16 + lrow;
#pragma unroll
      for (int r = 0; r < 8; ++r)
        op[(size_t)(rowb + r) * D_ + col] = accO[dt][r];
    }
  }
}

extern "C" void kernel_launch(void* const* d_in, const int* in_sizes, int n_in,
                              void* d_out, int out_size, void* d_ws, size_t ws_size,
                              hipStream_t stream) {
  (void)in_sizes; (void)n_in; (void)d_ws; (void)ws_size; (void)out_size;
  const float* a      = (const float*)d_in[0];
  const float* a_mask = (const float*)d_in[1];
  const float* bmat   = (const float*)d_in[2];
  const float* b_mask = (const float*)d_in[3];
  float* out = (float*)d_out;
  const size_t half = (size_t)B_ * L_ * D_;

  hipFuncSetAttribute((const void*)attn_fused_kernel,
                      hipFuncAttributeMaxDynamicSharedMemorySize, SMEM_BYTES);

  dim3 grid(B_ * (L_ / TQ));   // 512 workgroups
  // attended_a = softmax(A B^T + b_bias) B
  attn_fused_kernel<<<grid, 256, SMEM_BYTES, stream>>>(a, bmat, b_mask, out);
  // attended_b = softmax(B A^T + a_bias) A
  attn_fused_kernel<<<grid, 256, SMEM_BYTES, stream>>>(bmat, a, a_mask, out + half);
}
